// Recurrence_12352325944186
// MI455X (gfx1250) — matrix operand order repaired and checked by reading, not compiled, not run on hardware
//
#include <hip/hip_runtime.h>
#include <hip/hip_bf16.h>

// ---------------------------------------------------------------------------
// MI455X (gfx1250, wave32) implementation.
// All matmuls run through v_wmma_f32_16x16x32_f16 (f16 in, f32 accumulate).
// GEMM: 64x128 block tile, 8 waves, each wave owns a 32x32 tile (4 WMMA per
// K-chunk). B (weights) staged global->LDS with GLOBAL_LOAD_ASYNC_TO_LDS_B128
// (ASYNCcnt / s_wait_asynccnt), A staged through registers (ReLU applied on
// load). Double-buffered LDS overlaps the async DMA + A loads with WMMA.
// ---------------------------------------------------------------------------

typedef _Float16 h16;
typedef __attribute__((ext_vector_type(8)))  _Float16 v8h;
typedef __attribute__((ext_vector_type(16))) _Float16 v16h;
typedef __attribute__((ext_vector_type(8)))  float    v8f;

#define GSL(i, n) for (size_t i = (size_t)blockIdx.x * blockDim.x + threadIdx.x; \
                       i < (size_t)(n); i += (size_t)gridDim.x * blockDim.x)

static inline unsigned gblocks(size_t n) {
    size_t b = (n + 255) / 256;
    return (unsigned)(b > 65535 ? 65535 : b);
}

// ----------------------------- constants -----------------------------------
#define TT 16
#define NN 512
#define PP 16
#define OO 128
#define HH 1024
#define EE 512
#define LL 2
#define NVV 32
#define NOBS 64
#define ACT 17
#define DIN 81            // NOBS + ACT
#define STATE 6692
#define ROWS (TT*NN)      // 8192
#define INFV 100000000.0f

// offsets within STATE (cumsum of concat sizes)
#define OFF_VALUES 0
#define OFF_PROBS  2048
#define OFF_OPTS   4096
#define OFF_INDS   4112
#define OFF_ML     4128
#define OFF_ELOSS  4129
#define OFF_HF     4130
#define OFF_EMB2   6178
#define OFF_A      6690
#define OFF_VV     6691

// ----------------------------- WMMA GEMM -----------------------------------
// C[M,N] = act(A[M,K]) @ W[N,K]^T + bias ; A,W are f16 row-major (K contig).
// Block = 256 threads (8 waves). Block tile 64(M) x 128(N). Wave grid 2x4 of
// 32x32 wave tiles; each wave tile = 2x2 WMMA 16x16 tiles. K chunks of 32
// double-buffered through LDS; B side uses async global->LDS DMA.
#define TM 64
#define TN 128
#define TK 32
#define LDP 48   // padded LDS row (halves); keeps 16B fragment alignment

// async B-staging: each thread copies 32B (2 x b128) of W straight into LDS.
// INST_OFFSET is applied to BOTH the LDS and global addresses (ISA 10.7/15.18),
// so one base pair + offset:16 covers the thread's row segment.
__device__ __forceinline__ void async_copy_b32B(const h16* gsrc, h16* ldst) {
    const unsigned lds_addr = (unsigned)(unsigned long long)ldst;  // low 32b = LDS offset
    asm volatile(
        "global_load_async_to_lds_b128 %0, %1, off\n\t"
        "global_load_async_to_lds_b128 %0, %1, off offset:16"
        :: "v"(lds_addr), "v"(gsrc) : "memory");
}
__device__ __forceinline__ void async_wait0() {
    asm volatile("s_wait_asynccnt 0x0" ::: "memory");
}

__global__ __launch_bounds__(256)
void wmma_gemm_kernel(const h16* __restrict__ A, const h16* __restrict__ W,
                      const float* __restrict__ bias,
                      float* __restrict__ Cf, h16* __restrict__ Ch,
                      int M, int N, int K, int ldc, int relu_a)
{
    __shared__ h16 As[2][TM][LDP];
    __shared__ h16 Bs[2][TN][LDP];

    const int tid  = threadIdx.x;
    const int lane = tid & 31;
    const int wave = tid >> 5;
    const int mi   = wave >> 2;       // 0..1
    const int ni   = wave & 3;        // 0..3
    const int bm   = blockIdx.y * TM;
    const int bn   = blockIdx.x * TN;

    // cooperative staging: A 64x32 (8 halves/thread), B 128x32 (16 halves/thread)
    const int arow = tid >> 2, acol = (tid & 3) * 8;
    const int brow = tid >> 1, bcol = (tid & 1) * 16;

    const int half = lane >> 4;       // 0/1
    const int lm   = lane & 15;

    v8f acc00 = {}, acc01 = {}, acc10 = {}, acc11 = {};

    const int nchunks = K / TK;

    // ---- preload chunk 0 (B via async DMA, A via registers w/ ReLU) ----
    async_copy_b32B(W + (size_t)(bn + brow) * K + bcol, &Bs[0][brow][bcol]);
    v8h areg = *(const v8h*)(A + (size_t)(bm + arow) * K + acol);
    if (relu_a) {
        #pragma unroll
        for (int i = 0; i < 8; ++i) areg[i] = areg[i] > (h16)0 ? areg[i] : (h16)0;
    }
    *(v8h*)(&As[0][arow][acol]) = areg;
    async_wait0();
    __syncthreads();

    for (int c = 0; c < nchunks; ++c) {
        const int cur = c & 1, nxt = cur ^ 1;

        // ---- kick next chunk early: async B DMA + A global loads overlap WMMA ----
        if (c + 1 < nchunks) {
            const int k0 = (c + 1) * TK;
            async_copy_b32B(W + (size_t)(bn + brow) * K + k0 + bcol,
                            &Bs[nxt][brow][bcol]);
            areg = *(const v8h*)(A + (size_t)(bm + arow) * K + k0 + acol);
            if (relu_a) {
                #pragma unroll
                for (int i = 0; i < 8; ++i) areg[i] = areg[i] > (h16)0 ? areg[i] : (h16)0;
            }
            if (c + 2 < nchunks) {   // pull chunk c+2 toward the caches
                __builtin_prefetch(A + (size_t)(bm + arow) * K + k0 + TK + acol, 0, 0);
                __builtin_prefetch(W + (size_t)(bn + brow) * K + k0 + TK + bcol, 0, 0);
            }
        }

        // ---- fragments (A: 16-bit 16x32 layout, B: 32x16 layout) ----
        v8h alo0 = *(const v8h*)(&As[cur][mi * 32 + lm][half * 8]);
        v8h ahi0 = *(const v8h*)(&As[cur][mi * 32 + lm][16 + half * 8]);
        v16h af0 = __builtin_shufflevector(alo0, ahi0,
                    0,1,2,3,4,5,6,7,8,9,10,11,12,13,14,15);
        v8h alo1 = *(const v8h*)(&As[cur][mi * 32 + 16 + lm][half * 8]);
        v8h ahi1 = *(const v8h*)(&As[cur][mi * 32 + 16 + lm][16 + half * 8]);
        v16h af1 = __builtin_shufflevector(alo1, ahi1,
                    0,1,2,3,4,5,6,7,8,9,10,11,12,13,14,15);
        v8h blo0 = *(const v8h*)(&Bs[cur][ni * 32 + lm][half * 8]);
        v8h bhi0 = *(const v8h*)(&Bs[cur][ni * 32 + lm][16 + half * 8]);
        v16h bf0 = __builtin_shufflevector(blo0, bhi0,
                    0,1,2,3,4,5,6,7,8,9,10,11,12,13,14,15);
        v8h blo1 = *(const v8h*)(&Bs[cur][ni * 32 + 16 + lm][half * 8]);
        v8h bhi1 = *(const v8h*)(&Bs[cur][ni * 32 + 16 + lm][16 + half * 8]);
        v16h bf1 = __builtin_shufflevector(blo1, bhi1,
                    0,1,2,3,4,5,6,7,8,9,10,11,12,13,14,15);

        acc00 = __builtin_amdgcn_wmma_f32_16x16x32_f16(false, af0, false, bf0, (short)0, acc00, false, false);
        acc01 = __builtin_amdgcn_wmma_f32_16x16x32_f16(false, af0, false, bf1, (short)0, acc01, false, false);
        acc10 = __builtin_amdgcn_wmma_f32_16x16x32_f16(false, af1, false, bf0, (short)0, acc10, false, false);
        acc11 = __builtin_amdgcn_wmma_f32_16x16x32_f16(false, af1, false, bf1, (short)0, acc11, false, false);

        // ---- stage next A chunk; drain async B DMA before the barrier ----
        if (c + 1 < nchunks) {
            *(v8h*)(&As[nxt][arow][acol]) = areg;
        }
        async_wait0();
        __syncthreads();
    }

    // ---- epilogue: C layout — lane half selects M 0..7/8..15, n = lane%16 ----
    #pragma unroll
    for (int m0 = 0; m0 < 2; ++m0) {
        #pragma unroll
        for (int n0 = 0; n0 < 2; ++n0) {
            const v8f acc = m0 ? (n0 ? acc11 : acc10) : (n0 ? acc01 : acc00);
            const int n = bn + ni * 32 + n0 * 16 + lm;
            const float bv = bias ? bias[n] : 0.0f;
            #pragma unroll
            for (int r = 0; r < 8; ++r) {
                const int m = bm + mi * 32 + m0 * 16 + half * 8 + r;
                const float v = acc[r] + bv;
                if (Cf) Cf[(size_t)m * ldc + n] = v;
                if (Ch) Ch[(size_t)m * ldc + n] = (h16)v;
            }
        }
    }
}

// ----------------------------- small kernels --------------------------------
__global__ void f32_to_f16_kernel(const float* __restrict__ in, h16* __restrict__ out, size_t n) {
    GSL(i, n) out[i] = (h16)in[i];
}
__global__ void fill_f32_kernel(float* p, float v, size_t n) { GSL(i, n) p[i] = v; }
__global__ void fill_i32_kernel(int* p, int v, size_t n)     { GSL(i, n) p[i] = v; }
__global__ void copy_f32_kernel(const float* a, float* b, size_t n) { GSL(i, n) b[i] = a[i]; }

// x[r, k*32+c] = emb_obs[obs[r,k]*32 + c]  (raw f16; relu applied in GEMM)
__global__ void gather_embed_kernel(const float* __restrict__ inputs,
                                    const float* __restrict__ emb_obs,
                                    h16* __restrict__ x) {
    GSL(i, (size_t)ROWS * 2048) {
        const size_t r = i >> 11;
        const int rem = (int)(i & 2047);
        const int k = rem >> 5, c = rem & 31;
        const int o = (int)inputs[r * DIN + k];
        x[i] = (h16)emb_obs[o * NVV + c];
    }
}

// states[:, j] = new_state ; indices[n, J[n]] = j ; reset push_any
__global__ void set_state_kernel(const float* __restrict__ ns,
                                 float* __restrict__ states, h16* __restrict__ statesh,
                                 const int* __restrict__ J, int* __restrict__ indices,
                                 int* __restrict__ push_any, int j) {
    GSL(i, (size_t)NN * EE) {
        const int n = (int)(i >> 9), e = (int)(i & 511);
        const float v = ns[i];
        states [((size_t)n * PP + j) * EE + e] = v;
        statesh[((size_t)n * PP + j) * EE + e] = (h16)v;
        if (e == 0) { indices[n * PP + J[n]] = j; if (n == 0) *push_any = 0; }
    }
}

__global__ void gather_xj_kernel(const h16* __restrict__ statesh, const int* __restrict__ J,
                                 h16* __restrict__ xj) {
    GSL(i, (size_t)NN * EE) {
        const int n = (int)(i >> 9), e = (int)(i & 511);
        xj[i] = statesh[((size_t)n * PP + J[n]) * EE + e];
    }
}

// sharp[n] = relu(xj[n]) . W_sharp + b_sharp   (512 blocks x 32 lanes)
__global__ void sharp_kernel(const h16* __restrict__ xj, const float* __restrict__ Ws,
                             const float* __restrict__ bs, float* __restrict__ sharp) {
    const int n = blockIdx.x, lane = threadIdx.x;
    float s = 0.0f;
    for (int e = lane; e < EE; e += 32) {
        float x = (float)xj[(size_t)n * EE + e];
        x = x > 0.0f ? x : 0.0f;
        s += x * Ws[e];
    }
    for (int off = 16; off; off >>= 1) s += __shfl_down(s, off);
    if (lane == 0) sharp[n] = s + bs[0];
}

__device__ inline float hash_unit(unsigned x) {
    x ^= x >> 16; x *= 0x7feb352du; x ^= x >> 15; x *= 0x846ca68bu; x ^= x >> 16;
    return (float)(x >> 8) * (1.0f / 16777216.0f) + 1e-7f;
}

// hash-Gumbel categorical sample + logits/options/push bookkeeping
__global__ void sample_kernel(float* __restrict__ logits, const float* __restrict__ values,
                              const float* __restrict__ sharp, int* __restrict__ options,
                              const int* __restrict__ J, int* __restrict__ push,
                              int* __restrict__ push_any, int j) {
    const int n = blockIdx.x * blockDim.x + threadIdx.x;
    if (n >= NN) return;
    const int Jn = J[n];
    const float* lJ = logits + ((size_t)n * PP + Jn) * OO;
    bool fresh = true;
    for (int o = 0; o < OO; ++o) fresh = fresh && (lJ[o] == 0.0f);
    const float sh = sharp[n];
    const float* vj = values + ((size_t)n * PP + j) * OO;

    int best = 0; float bestv = -1e30f;
    for (int o = 0; o < OO; ++o) {
        const float l = fresh ? sh * vj[o] : lJ[o];
        const unsigned seed = (unsigned)(j * 131071) ^ (unsigned)(n * 4099) ^ (unsigned)(o * 2654435769u);
        const float g = -__logf(-__logf(hash_unit(seed)));
        const float c = l + g;
        if (c > bestv) { bestv = c; best = o; }
    }
    float* lout = logits + ((size_t)n * PP + j) * OO;
    for (int o = 0; o < OO; ++o) {
        const float l = fresh ? sh * vj[o] : lJ[o];
        lout[o] = l - (o == best ? INFV : 0.0f);
    }
    options[n * PP + j]  = best;
    options[n * PP + Jn] = best;
    const int p = vj[best] > 0.0f ? 1 : 0;
    push[n] = p;
    if (p) atomicOr(push_any, 1);
}

// minput = concat(states[n,J], emb_opt[options[n,J]])
__global__ void build_minput_p_kernel(const h16* __restrict__ statesh,
                                      const float* __restrict__ emb_opt,
                                      const int* __restrict__ options,
                                      const int* __restrict__ J,
                                      h16* __restrict__ minput) {
    GSL(i, (size_t)NN * (EE + OO)) {
        const int n = (int)(i / (EE + OO)), c = (int)(i % (EE + OO));
        const int Jn = J[n];
        if (c < EE) minput[i] = statesh[((size_t)n * PP + Jn) * EE + c];
        else {
            const int opt = options[n * PP + Jn];
            minput[i] = (h16)emb_opt[opt * OO + (c - EE)];
        }
    }
}

// hJ gather (hidden layout [n][P][L][H])
__global__ void gather_h_kernel(const float* __restrict__ hidden, const int* __restrict__ J,
                                float* hj0, h16* hj0h, float* hj1, h16* hj1h) {
    GSL(i, (size_t)NN * HH) {
        const int n = (int)(i >> 10), hh = (int)(i & 1023);
        const float* base = hidden + ((size_t)n * PP + J[n]) * (LL * HH);
        const float a = base[hh], b = base[HH + hh];
        hj0[i] = a; hj0h[i] = (h16)a;
        hj1[i] = b; hj1h[i] = (h16)b;
    }
}

__global__ void gru_pointwise_kernel(const float* __restrict__ gi, const float* __restrict__ gh,
                                     const float* __restrict__ hprev,
                                     float* __restrict__ hout, h16* __restrict__ houth) {
    GSL(i, (size_t)NN * HH) {
        const int n = (int)(i >> 10), hh = (int)(i & 1023);
        const size_t b = (size_t)n * (3 * HH);
        const float ir = gi[b + hh], iz = gi[b + HH + hh], in = gi[b + 2 * HH + hh];
        const float hr = gh[b + hh], hz = gh[b + HH + hh], hn = gh[b + 2 * HH + hh];
        const float r = 1.0f / (1.0f + __expf(-(ir + hr)));
        const float z = 1.0f / (1.0f + __expf(-(iz + hz)));
        const float nn = tanhf(in + r * hn);
        const float h = (1.0f - z) * nn + z * hprev[i];
        hout[i] = h;
        houth[i] = (h16)h;
    }
}

__global__ void update_hidden_kernel(const float* __restrict__ h0, const float* __restrict__ h1,
                                     const int* __restrict__ push_any,
                                     float* __restrict__ hidden, int j) {
    if (!*push_any) return;
    GSL(i, (size_t)NN * LL * HH) {
        const int n = (int)(i >> 11);
        const int rem = (int)(i & 2047);
        const int l = rem >> 10, hh = rem & 1023;
        hidden[((size_t)n * PP + j) * (LL * HH) + l * HH + hh] =
            l ? h1[(size_t)n * HH + hh] : h0[(size_t)n * HH + hh];
    }
}

__global__ void update_state_J_kernel(const float* __restrict__ cand,
                                      const float* __restrict__ states,
                                      const int* __restrict__ push,
                                      float* __restrict__ ns, int* __restrict__ J, int j) {
    GSL(i, (size_t)NN * EE) {
        const int n = (int)(i >> 9), e = (int)(i & 511);
        ns[i] = push[n] ? cand[i] : states[((size_t)n * PP + j) * EE + e];
        if (e == 0) {
            int Jn = J[n];
            Jn = push[n] ? min(Jn + 1, PP - 1) : max(Jn - 1, 0);
            J[n] = Jn;
        }
    }
}

__global__ void softmax_kernel(const float* __restrict__ logits, float* __restrict__ probs) {
    const int row = blockIdx.x * blockDim.x + threadIdx.x;
    if (row >= NN * PP) return;
    const float* l = logits + (size_t)row * OO;
    float* p = probs + (size_t)row * OO;
    float m = -1e30f;
    for (int o = 0; o < OO; ++o) m = fmaxf(m, l[o]);
    float s = 0.0f;
    for (int o = 0; o < OO; ++o) { const float e = __expf(l[o] - m); p[o] = e; s += e; }
    const float inv = 1.0f / s;
    for (int o = 0; o < OO; ++o) p[o] *= inv;
}

__global__ void scan_init_kernel(const float* __restrict__ rnn,
                                 float* sh0, h16* sh0h, float* sh1, h16* sh1h) {
    GSL(i, (size_t)NN * HH) {
        const int n = (int)(i >> 10), hh = (int)(i & 1023);
        const float a = rnn[(size_t)n * STATE + OFF_HF + hh * LL + 0];
        const float b = rnn[(size_t)n * STATE + OFF_HF + hh * LL + 1];
        sh0[i] = a; sh0h[i] = (h16)a;
        sh1[i] = b; sh1h[i] = (h16)b;
    }
}

__global__ void scan_meta_kernel(const int* __restrict__ indices, const int* __restrict__ options,
                                 const float* __restrict__ values,
                                 int* __restrict__ opt_buf, float* __restrict__ out, int t) {
    const int n = blockIdx.x * blockDim.x + threadIdx.x;
    if (n >= NN) return;
    const int Jt = indices[n * PP + t];
    const int opt = options[n * PP + Jt];
    opt_buf[n] = opt;
    const size_t o = ((size_t)t * NN + n) * STATE;
    out[o + OFF_A]  = (float)opt;
    out[o + OFF_VV] = values[((size_t)n * PP + Jt) * OO + opt];
}

__global__ void build_minput_scan_kernel(const float* __restrict__ rnn,
                                         const float* __restrict__ emb2_seq,
                                         const float* __restrict__ emb_opt,
                                         const int* __restrict__ opt_buf,
                                         h16* __restrict__ minput, int t) {
    GSL(i, (size_t)NN * (EE + OO)) {
        const int n = (int)(i / (EE + OO)), c = (int)(i % (EE + OO));
        if (c < EE) {
            const float v = (t == 0)
                ? rnn[(size_t)n * STATE + OFF_EMB2 + c]
                : emb2_seq[((size_t)(t - 1) * NN + n) * EE + c];
            minput[i] = (h16)v;
        } else {
            minput[i] = (h16)emb_opt[opt_buf[n] * OO + (c - EE)];
        }
    }
}

__global__ void ml_kernel(const float* __restrict__ e2, const float* __restrict__ emb2_seq,
                          float* __restrict__ out, int t) {
    const int n = blockIdx.x, lane = threadIdx.x;
    const float* tgt = emb2_seq + ((size_t)t * NN + n) * EE;
    float s = 0.0f;
    for (int e = lane; e < EE; e += 32) {
        const float d = e2[(size_t)n * EE + e] - tgt[e];
        s += d * d;
    }
    for (int off = 16; off; off >>= 1) s += __shfl_down(s, off);
    if (lane == 0) out[((size_t)t * NN + n) * STATE + OFF_ML] = s * (1.0f / EE);
}

__global__ void hf_kernel(const float* __restrict__ sh0, const float* __restrict__ sh1,
                          float* __restrict__ out, int t) {
    GSL(i, (size_t)NN * HH) {
        const int n = (int)(i >> 10), hh = (int)(i & 1023);
        const size_t o = ((size_t)t * NN + n) * STATE + OFF_HF;
        out[o + hh * LL + 0] = sh0[i];
        out[o + hh * LL + 1] = sh1[i];
    }
}

__global__ void broadcast_kernel(const float* __restrict__ values, const float* __restrict__ probs,
                                 const int* __restrict__ options, const int* __restrict__ indices,
                                 const float* __restrict__ rnn, const float* __restrict__ emb2_seq,
                                 float* __restrict__ out) {
    GSL(idx, (size_t)TT * NN * STATE) {
        const int t = (int)(idx / ((size_t)NN * STATE));
        const int rem = (int)(idx % ((size_t)NN * STATE));
        const int n = rem / STATE, c = rem % STATE;
        float v;
        if      (c < OFF_PROBS)  v = values[(size_t)n * (PP * OO) + c];
        else if (c < OFF_OPTS)   v = probs[(size_t)n * (PP * OO) + (c - OFF_PROBS)];
        else if (c < OFF_INDS)   v = (float)options[n * PP + (c - OFF_OPTS)];
        else if (c < OFF_ML)     v = (float)indices[n * PP + (c - OFF_INDS)];
        else if (c == OFF_ELOSS) v = rnn[(size_t)n * STATE + OFF_ELOSS];
        else if (c >= OFF_EMB2 && c < OFF_A)
            v = emb2_seq[((size_t)t * NN + n) * EE + (c - OFF_EMB2)];
        else continue;   // scan-written fields (ml, hf, a, vv)
        out[idx] = v;
    }
}

__global__ void copy_last_kernel(float* __restrict__ out) {
    GSL(i, (size_t)NN * STATE) {
        out[(size_t)TT * NN * STATE + i] = out[(size_t)(TT - 1) * NN * STATE + i];
    }
}

// ----------------------------- host side -----------------------------------
static void gemm(hipStream_t s, const h16* A, const h16* W, const float* bias,
                 float* Cf, h16* Ch, int M, int N, int K, int ldc, int relu_a) {
    dim3 grid(N / TN, M / TM);
    wmma_gemm_kernel<<<grid, 256, 0, s>>>(A, W, bias, Cf, Ch, M, N, K, ldc, relu_a);
}

static void cvt(hipStream_t s, const float* in, h16* out, size_t n) {
    f32_to_f16_kernel<<<gblocks(n), 256, 0, s>>>(in, out, n);
}

extern "C" void kernel_launch(void* const* d_in, const int* in_sizes, int n_in,
                              void* d_out, int out_size, void* d_ws, size_t ws_size,
                              hipStream_t stream) {
    (void)in_sizes; (void)n_in; (void)out_size; (void)ws_size;

    const float* inputs  = (const float*)d_in[0];
    const float* rnn     = (const float*)d_in[1];
    const float* emb_obs = (const float*)d_in[2];
    const float* W_e0    = (const float*)d_in[3];
    const float* b_e0    = (const float*)d_in[4];
    const float* W_e1    = (const float*)d_in[5];
    const float* b_e1    = (const float*)d_in[6];
    const float* W_emb2  = (const float*)d_in[7];
    const float* b_emb2  = (const float*)d_in[8];
    const float* W_sharp = (const float*)d_in[9];
    const float* b_sharp = (const float*)d_in[10];
    const float* W_crit  = (const float*)d_in[11];
    const float* b_crit  = (const float*)d_in[12];
    const float* emb_opt = (const float*)d_in[13];
    const float* Wih0    = (const float*)d_in[14];
    const float* Whh0    = (const float*)d_in[15];
    const float* bih0    = (const float*)d_in[16];
    const float* bhh0    = (const float*)d_in[17];
    const float* Wih1    = (const float*)d_in[18];
    const float* Whh1    = (const float*)d_in[19];
    const float* bih1    = (const float*)d_in[20];
    const float* bhh1    = (const float*)d_in[21];
    float* out = (float*)d_out;

    // ---- workspace carve ----
    char* wp = (char*)d_ws;
    auto carve = [&](size_t bytes) -> void* {
        void* r = (void*)wp;
        wp += (bytes + 255) & ~(size_t)255;
        return r;
    };
    // f16 weights
    h16* wW_e0   = (h16*)carve(sizeof(h16) * 1024 * 2048);
    h16* wW_e1   = (h16*)carve(sizeof(h16) * 1024 * 1024);
    h16* wW_emb2 = (h16*)carve(sizeof(h16) * 512 * 1024);
    h16* wW_crit = (h16*)carve(sizeof(h16) * 128 * 512);
    h16* wWih0   = (h16*)carve(sizeof(h16) * 3072 * 640);
    h16* wWhh0   = (h16*)carve(sizeof(h16) * 3072 * 1024);
    h16* wWih1   = (h16*)carve(sizeof(h16) * 3072 * 1024);
    h16* wWhh1   = (h16*)carve(sizeof(h16) * 3072 * 1024);
    // f16 activations
    h16* xf16    = (h16*)carve(sizeof(h16) * (size_t)ROWS * 2048);
    h16* h1f16   = (h16*)carve(sizeof(h16) * (size_t)ROWS * HH);
    h16* emb1h   = (h16*)carve(sizeof(h16) * (size_t)ROWS * HH);
    h16* statesh = (h16*)carve(sizeof(h16) * (size_t)NN * PP * EE);
    h16* xjh     = (h16*)carve(sizeof(h16) * (size_t)NN * EE);
    h16* minputh = (h16*)carve(sizeof(h16) * (size_t)NN * (EE + OO));
    h16* hj0h    = (h16*)carve(sizeof(h16) * (size_t)NN * HH);
    h16* hj1h    = (h16*)carve(sizeof(h16) * (size_t)NN * HH);
    h16* hn0h    = (h16*)carve(sizeof(h16) * (size_t)NN * HH);
    h16* hn1h    = (h16*)carve(sizeof(h16) * (size_t)NN * HH);
    h16* sh0h    = (h16*)carve(sizeof(h16) * (size_t)NN * HH);
    h16* sh1h    = (h16*)carve(sizeof(h16) * (size_t)NN * HH);
    // f32 buffers
    float* emb2_seq = (float*)carve(sizeof(float) * (size_t)ROWS * EE);
    float* states   = (float*)carve(sizeof(float) * (size_t)NN * PP * EE);
    float* values   = (float*)carve(sizeof(float) * (size_t)NN * PP * OO);
    float* logits   = (float*)carve(sizeof(float) * (size_t)NN * PP * OO);
    float* probs    = (float*)carve(sizeof(float) * (size_t)NN * PP * OO);
    float* hidden   = (float*)carve(sizeof(float) * (size_t)NN * PP * LL * HH);
    float* gi       = (float*)carve(sizeof(float) * (size_t)NN * 3 * HH);
    float* gh       = (float*)carve(sizeof(float) * (size_t)NN * 3 * HH);
    float* hj0      = (float*)carve(sizeof(float) * (size_t)NN * HH);
    float* hj1      = (float*)carve(sizeof(float) * (size_t)NN * HH);
    float* hn0      = (float*)carve(sizeof(float) * (size_t)NN * HH);
    float* hn1      = (float*)carve(sizeof(float) * (size_t)NN * HH);
    float* sh0      = (float*)carve(sizeof(float) * (size_t)NN * HH);
    float* sh1      = (float*)carve(sizeof(float) * (size_t)NN * HH);
    float* cand     = (float*)carve(sizeof(float) * (size_t)NN * EE);
    float* nstate   = (float*)carve(sizeof(float) * (size_t)NN * EE);
    float* e2       = (float*)carve(sizeof(float) * (size_t)NN * EE);
    float* sharp    = (float*)carve(sizeof(float) * NN);
    int* J        = (int*)carve(sizeof(int) * NN);
    int* push     = (int*)carve(sizeof(int) * NN);
    int* push_any = (int*)carve(sizeof(int) * 16);
    int* opt_buf  = (int*)carve(sizeof(int) * NN);
    int* options  = (int*)carve(sizeof(int) * NN * PP);
    int* indices  = (int*)carve(sizeof(int) * NN * PP);

    // ---- weight conversion ----
    cvt(stream, W_e0,   wW_e0,   (size_t)1024 * 2048);
    cvt(stream, W_e1,   wW_e1,   (size_t)1024 * 1024);
    cvt(stream, W_emb2, wW_emb2, (size_t)512 * 1024);
    cvt(stream, W_crit, wW_crit, (size_t)128 * 512);
    cvt(stream, Wih0,   wWih0,   (size_t)3072 * 640);
    cvt(stream, Whh0,   wWhh0,   (size_t)3072 * 1024);
    cvt(stream, Wih1,   wWih1,   (size_t)3072 * 1024);
    cvt(stream, Whh1,   wWhh1,   (size_t)3072 * 1024);

    // ---- phase 1: embedding + encoder MLP + emb2_seq ----
    gather_embed_kernel<<<gblocks((size_t)ROWS * 2048), 256, 0, stream>>>(inputs, emb_obs, xf16);
    gemm(stream, xf16,  wW_e0,   b_e0,   nullptr, h1f16, ROWS, HH, 2048, HH, 1);
    gemm(stream, h1f16, wW_e1,   b_e1,   nullptr, emb1h, ROWS, HH, HH,   HH, 1);
    gemm(stream, emb1h, wW_emb2, b_emb2, emb2_seq, nullptr, ROWS, EE, HH, EE, 1);

    // ---- init recurrent state ----
    fill_f32_kernel<<<gblocks((size_t)NN * PP * OO), 256, 0, stream>>>(logits, 0.0f, (size_t)NN * PP * OO);
    fill_f32_kernel<<<gblocks((size_t)NN * PP * LL * HH), 256, 0, stream>>>(hidden, 0.0f, (size_t)NN * PP * LL * HH);
    fill_i32_kernel<<<gblocks((size_t)NN * PP), 256, 0, stream>>>(indices, 0, (size_t)NN * PP);
    fill_i32_kernel<<<gblocks((size_t)NN * PP), 256, 0, stream>>>(options, -1, (size_t)NN * PP);
    fill_i32_kernel<<<gblocks(NN), 256, 0, stream>>>(J, 0, NN);
    copy_f32_kernel<<<gblocks((size_t)NN * EE), 256, 0, stream>>>(emb2_seq, nstate, (size_t)NN * EE);

    // ---- phase 2: P-step option stack ----
    for (int j = 0; j < PP; ++j) {
        set_state_kernel<<<gblocks((size_t)NN * EE), 256, 0, stream>>>(nstate, states, statesh, J, indices, push_any, j);
        gather_xj_kernel<<<gblocks((size_t)NN * EE), 256, 0, stream>>>(statesh, J, xjh);
        gemm(stream, xjh, wW_crit, b_crit, values + (size_t)j * OO, nullptr, NN, OO, EE, PP * OO, 1);
        sharp_kernel<<<NN, 32, 0, stream>>>(xjh, W_sharp, b_sharp, sharp);
        sample_kernel<<<(NN + 255) / 256, 256, 0, stream>>>(logits, values, sharp, options, J, push, push_any, j);
        build_minput_p_kernel<<<gblocks((size_t)NN * (EE + OO)), 256, 0, stream>>>(statesh, emb_opt, options, J, minputh);
        gather_h_kernel<<<gblocks((size_t)NN * HH), 256, 0, stream>>>(hidden, J, hj0, hj0h, hj1, hj1h);
        // GRU layer 0
        gemm(stream, minputh, wWih0, bih0, gi, nullptr, NN, 3 * HH, EE + OO, 3 * HH, 0);
        gemm(stream, hj0h,    wWhh0, bhh0, gh, nullptr, NN, 3 * HH, HH,      3 * HH, 0);
        gru_pointwise_kernel<<<gblocks((size_t)NN * HH), 256, 0, stream>>>(gi, gh, hj0, hn0, hn0h);
        // GRU layer 1
        gemm(stream, hn0h, wWih1, bih1, gi, nullptr, NN, 3 * HH, HH, 3 * HH, 0);
        gemm(stream, hj1h, wWhh1, bhh1, gh, nullptr, NN, 3 * HH, HH, 3 * HH, 0);
        gru_pointwise_kernel<<<gblocks((size_t)NN * HH), 256, 0, stream>>>(gi, gh, hj1, hn1, hn1h);
        // embed2(mout)
        gemm(stream, hn1h, wW_emb2, b_emb2, cand, nullptr, NN, EE, HH, EE, 1);
        update_hidden_kernel<<<gblocks((size_t)NN * LL * HH), 256, 0, stream>>>(hn0, hn1, push_any, hidden, j);
        update_state_J_kernel<<<gblocks((size_t)NN * EE), 256, 0, stream>>>(cand, states, push, nstate, J, j);
    }

    softmax_kernel<<<(NN * PP + 255) / 256, 256, 0, stream>>>(logits, probs);
    broadcast_kernel<<<gblocks((size_t)TT * NN * STATE), 256, 0, stream>>>(values, probs, options, indices, rnn, emb2_seq, out);

    // ---- phase 3: T-step scan ----
    scan_init_kernel<<<gblocks((size_t)NN * HH), 256, 0, stream>>>(rnn, sh0, sh0h, sh1, sh1h);
    for (int t = 0; t < TT; ++t) {
        scan_meta_kernel<<<(NN + 255) / 256, 256, 0, stream>>>(indices, options, values, opt_buf, out, t);
        build_minput_scan_kernel<<<gblocks((size_t)NN * (EE + OO)), 256, 0, stream>>>(rnn, emb2_seq, emb_opt, opt_buf, minputh, t);
        // GRU layer 0 (state sh0)
        gemm(stream, minputh, wWih0, bih0, gi, nullptr, NN, 3 * HH, EE + OO, 3 * HH, 0);
        gemm(stream, sh0h,    wWhh0, bhh0, gh, nullptr, NN, 3 * HH, HH,      3 * HH, 0);
        gru_pointwise_kernel<<<gblocks((size_t)NN * HH), 256, 0, stream>>>(gi, gh, sh0, sh0, sh0h);
        // GRU layer 1 (state sh1)
        gemm(stream, sh0h, wWih1, bih1, gi, nullptr, NN, 3 * HH, HH, 3 * HH, 0);
        gemm(stream, sh1h, wWhh1, bhh1, gh, nullptr, NN, 3 * HH, HH, 3 * HH, 0);
        gru_pointwise_kernel<<<gblocks((size_t)NN * HH), 256, 0, stream>>>(gi, gh, sh1, sh1, sh1h);
        // embed2(mout) + epilogues
        gemm(stream, sh1h, wW_emb2, b_emb2, e2, nullptr, NN, EE, HH, EE, 1);
        ml_kernel<<<NN, 32, 0, stream>>>(e2, emb2_seq, out, t);
        hf_kernel<<<gblocks((size_t)NN * HH), 256, 0, stream>>>(sh0, sh1, out, t);
    }

    // ---- hx[-1:] duplicate ----
    copy_last_kernel<<<gblocks((size_t)NN * STATE), 256, 0, stream>>>(out);
}
